// GraphConv_28905129902721
// MI455X (gfx1250) — matrix operand order repaired and checked by reading, not compiled, not run on hardware
//
#include <hip/hip_runtime.h>
#include <hip/hip_bf16.h>

// GCN forward for MI455X (gfx1250, wave32).
// Dense layers: v_wmma_f32_16x16x32_bf16 (f32 accumulate), weights in LDS bf16.
// SPMM: CSR-by-destination built once per call (int atomics only), then pure
// gather — one wave per node, no float atomics, no AGG zeroing. h (51MB) is
// L2-resident (192MB), so gathers stay on-chip. BN via two-pass reduction.

#define DH 128

typedef __attribute__((ext_vector_type(16))) __bf16 v16bf;
typedef __attribute__((ext_vector_type(8)))  float  v8f;

__device__ inline void atomicAddF(float* p, float v) {
  __hip_atomic_fetch_add(p, v, __ATOMIC_RELAXED, __HIP_MEMORY_SCOPE_AGENT);
}
__device__ inline int atomicAddI(int* p, int v) {
  return __hip_atomic_fetch_add(p, v, __ATOMIC_RELAXED, __HIP_MEMORY_SCOPE_AGENT);
}

// ---- WMMA tile loaders -----------------------------------------------------
// A (16x32 bf16, M x K): lanes 0-15 = rows, K {0..15};
// lanes 16-31 same rows, K {16..31}. Convert fp32 -> bf16 on the fly.
__device__ inline v16bf load_a_f32(const float* __restrict__ Abase, int rows, int lane) {
  int m = lane & 15;
  if (m >= rows) m = rows - 1;          // clamp; result masked at store
  const int kb = (lane >> 4) << 4;      // 0 or 16
  const float* p = Abase + (size_t)m * DH + kb;
  float4 f0 = *(const float4*)(p);
  float4 f1 = *(const float4*)(p + 4);
  float4 f2 = *(const float4*)(p + 8);
  float4 f3 = *(const float4*)(p + 12);
  v16bf a;
  a[0]  = (__bf16)f0.x; a[1]  = (__bf16)f0.y; a[2]  = (__bf16)f0.z; a[3]  = (__bf16)f0.w;
  a[4]  = (__bf16)f1.x; a[5]  = (__bf16)f1.y; a[6]  = (__bf16)f1.z; a[7]  = (__bf16)f1.w;
  a[8]  = (__bf16)f2.x; a[9]  = (__bf16)f2.y; a[10] = (__bf16)f2.z; a[11] = (__bf16)f2.w;
  a[12] = (__bf16)f3.x; a[13] = (__bf16)f3.y; a[14] = (__bf16)f3.z; a[15] = (__bf16)f3.w;
  return a;
}

// B (32x16 bf16, K x N): lane n = column, lanes 0-15 hold K=k0..k0+15
// contiguous, lanes 16-31 hold K=k0+16..k0+31. W row-major [C][K] -> each lane
// reads 16 contiguous bf16 (32B) from LDS.
__device__ inline v16bf load_b_lds(const __bf16* Wl, int c0, int k0, int lane) {
  const int c  = c0 + (lane & 15);
  const int kb = k0 + ((lane >> 4) << 4);   // 0 or 16
  const __bf16* p = Wl + c * DH + kb;
  v16bf b;
  __builtin_memcpy(&b, p, 32);              // 2x ds_load_b128
  return b;
}

// ---- GEMM: Out[N][CREAL] = A[N][128] @ W[CREAL][128]^T + bias --------------
template <int CPAD, int CREAL>
__global__ __launch_bounds__(256) void gemm_kernel(
    const float* __restrict__ A, const float* __restrict__ W,
    const float* __restrict__ bias, float* __restrict__ Out, int N) {
  __shared__ __bf16 Wl[CPAD * DH];
  __shared__ float  biasl[CPAD];
  for (int i = threadIdx.x; i < CPAD * DH; i += blockDim.x) {
    int c = i >> 7, k = i & (DH - 1);
    Wl[i] = (c < CREAL) ? (__bf16)W[c * DH + k] : (__bf16)0.f;
  }
  for (int i = threadIdx.x; i < CPAD; i += blockDim.x)
    biasl[i] = (i < CREAL) ? bias[i] : 0.f;
  __syncthreads();

  const int lane  = threadIdx.x & 31;
  const int wave  = threadIdx.x >> 5;
  const int wpb   = blockDim.x >> 5;
  const int ntile = (N + 15) >> 4;
  constexpr int NCT = CPAD / 16;

  for (int tile = blockIdx.x * wpb + wave; tile < ntile; tile += gridDim.x * wpb) {
    const int m0   = tile << 4;
    const int rows = (N - m0 < 16) ? (N - m0) : 16;
    v8f acc[NCT];
#pragma unroll
    for (int t = 0; t < NCT; ++t)
#pragma unroll
      for (int j = 0; j < 8; ++j) acc[t][j] = 0.f;

#pragma unroll
    for (int k0 = 0; k0 < DH; k0 += 32) {
      v16bf a = load_a_f32(A + (size_t)m0 * DH + k0, rows, lane);
#pragma unroll
      for (int ct = 0; ct < NCT; ++ct) {
        v16bf b = load_b_lds(Wl, ct * 16, k0, lane);
        acc[ct] = __builtin_amdgcn_wmma_f32_16x16x32_bf16(
            false, a, false, b, (short)0, acc[ct], false, false);
      }
    }
    // C/D layout: VGPR v = row v (lanes 0-15) / row v+8 (lanes 16-31), col = lane&15
    const int colIn = lane & 15;
    const int rAdd  = (lane >> 4) << 3;
#pragma unroll
    for (int ct = 0; ct < NCT; ++ct) {
      const int c = ct * 16 + colIn;
      if (c < CREAL) {
        const float bv = biasl[c];
#pragma unroll
        for (int v = 0; v < 8; ++v) {
          const int m = v + rAdd;
          if (m < rows) Out[(size_t)(m0 + m) * CREAL + c] = acc[ct][v] + bv;
        }
      }
    }
  }
}

// ---- CSR build --------------------------------------------------------------
__global__ __launch_bounds__(256) void zero_kernel(float* p, long long n) {
  long long i = (long long)blockIdx.x * blockDim.x + threadIdx.x;
  long long s = (long long)gridDim.x * blockDim.x;
  for (; i < n; i += s) p[i] = 0.f;
}

__global__ __launch_bounds__(256) void count_kernel(const int* __restrict__ col,
                                                    int* __restrict__ cnt, int E) {
  int e = blockIdx.x * blockDim.x + threadIdx.x;
  if (e < E) atomicAddI(&cnt[col[e]], 1);
}

// single-workgroup exclusive scan: offs[i] = sum(cnt[0..i)), cursor = copy
__global__ __launch_bounds__(1024) void scan_kernel(const int* __restrict__ cnt,
                                                    int* __restrict__ offs,
                                                    int* __restrict__ cursor, int N) {
  __shared__ int s[1024];
  __shared__ int carry;
  if (threadIdx.x == 0) carry = 0;
  __syncthreads();
  for (int base = 0; base < N; base += 1024) {
    const int i = base + (int)threadIdx.x;
    const int v = (i < N) ? cnt[i] : 0;
    s[threadIdx.x] = v;
    __syncthreads();
    for (int off = 1; off < 1024; off <<= 1) {
      int t = (threadIdx.x >= (unsigned)off) ? s[threadIdx.x - off] : 0;
      __syncthreads();
      s[threadIdx.x] += t;
      __syncthreads();
    }
    const int bc = carry;                // stable: last write was behind a barrier
    if (i < N) {
      const int excl = bc + s[threadIdx.x] - v;
      offs[i] = excl;
      cursor[i] = excl;
    }
    __syncthreads();
    if (threadIdx.x == 0) carry = bc + s[1023];
    __syncthreads();
  }
  if (threadIdx.x == 0) offs[N] = carry;
}

__global__ __launch_bounds__(256) void dinv_kernel(const int* __restrict__ cnt,
                                                   float* __restrict__ dinv, int N) {
  int i = blockIdx.x * blockDim.x + threadIdx.x;
  if (i < N) {
    float d = (float)cnt[i];
    dinv[i] = (d > 0.f) ? rsqrtf(d) : 0.f;
  }
}

__global__ __launch_bounds__(256) void fill_kernel(const int* __restrict__ rowI,
                                                   const int* __restrict__ colI,
                                                   int* __restrict__ cursor,
                                                   int* __restrict__ srcIdx, int E) {
  int e = blockIdx.x * blockDim.x + threadIdx.x;
  if (e < E) {
    int pos = atomicAddI(&cursor[colI[e]], 1);
    srcIdx[pos] = rowI[e];
  }
}

// ---- SPMM gather: one wave per destination node, no float atomics ----------
__global__ __launch_bounds__(256) void spmm_gather_kernel(
    const int* __restrict__ offs, const int* __restrict__ srcIdx,
    const float* __restrict__ dinv, const float* __restrict__ H,
    float* __restrict__ AGG, int N) {
  const int c = blockIdx.x * (blockDim.x >> 5) + (threadIdx.x >> 5);
  if (c >= N) return;
  const int lane  = threadIdx.x & 31;
  const int start = offs[c];
  const int end   = offs[c + 1];
  const float dc  = dinv[c];
  float4 acc; acc.x = acc.y = acc.z = acc.w = 0.f;
  for (int j = start; j < end; ++j) {
    const int   r = srcIdx[j];           // wave-uniform -> single L2 request
    const float v = dc * dinv[r];
    float4 hv = *((const float4*)(H + (size_t)r * DH) + lane);
    acc.x += v * hv.x; acc.y += v * hv.y; acc.z += v * hv.z; acc.w += v * hv.w;
  }
  *((float4*)(AGG + (size_t)c * DH) + lane) = acc;
}

// ---- BN ---------------------------------------------------------------------
__global__ __launch_bounds__(256) void stats_kernel(const float* __restrict__ H,
                                                    float* __restrict__ stats, int N) {
  __shared__ float s1[256], s2[256];
  const int c    = threadIdx.x & (DH - 1);
  const int half = threadIdx.x >> 7;
  float sum = 0.f, sq = 0.f;
  for (int i = blockIdx.x * 2 + half; i < N; i += gridDim.x * 2) {
    float v = H[(size_t)i * DH + c];
    sum += v; sq += v * v;
  }
  s1[threadIdx.x] = sum; s2[threadIdx.x] = sq;
  __syncthreads();
  if (threadIdx.x < DH) {
    atomicAddF(&stats[c],      s1[threadIdx.x] + s1[threadIdx.x + DH]);
    atomicAddF(&stats[DH + c], s2[threadIdx.x] + s2[threadIdx.x + DH]);
  }
}

__global__ __launch_bounds__(256) void bn_kernel(
    float* __restrict__ H, const float* __restrict__ stats,
    const float* __restrict__ gamma, const float* __restrict__ beta,
    const float* __restrict__ X0, float* __restrict__ X0out, int N) {
  long long total = (long long)N * DH;
  long long idx = (long long)blockIdx.x * blockDim.x + threadIdx.x;
  if (idx >= total) return;
  const int c = (int)(idx & (DH - 1));
  const float invN = 1.f / (float)N;
  const float mu   = stats[c] * invN;
  const float var  = stats[DH + c] * invN - mu * mu;
  const float is   = rsqrtf(var + 1e-5f);
  float v = (H[idx] - mu) * is * gamma[c] + beta[c];
  v = fmaxf(v, 0.f);
  if (X0) v += X0[idx];
  H[idx] = v;
  if (X0out) X0out[idx] = v;
}

// ---- host-side orchestration -----------------------------------------------
static inline int cdiv_i(int a, int b) { return (a + b - 1) / b; }
static inline size_t al4(size_t n) { return (n + 3) & ~(size_t)3; }

extern "C" void kernel_launch(void* const* d_in, const int* in_sizes, int n_in,
                              void* d_out, int out_size, void* d_ws, size_t ws_size,
                              hipStream_t stream) {
  const float* x        = (const float*)d_in[0];
  const int*   ei       = (const int*)  d_in[1];
  const float* fc_w     = (const float*)d_in[2];
  const float* fc_b     = (const float*)d_in[3];
  const float* conv_w   = (const float*)d_in[4];
  const float* conv_b   = (const float*)d_in[5];
  const float* bn_gamma = (const float*)d_in[6];
  const float* bn_beta  = (const float*)d_in[7];
  const float* cls_w    = (const float*)d_in[8];
  const float* cls_b    = (const float*)d_in[9];
  float*       out      = (float*)d_out;

  const int N = in_sizes[0] / DH;
  const int E = in_sizes[1] / 2;
  const int* rowI = ei;        // edge_index[0]
  const int* colI = ei + E;    // edge_index[1]

  // workspace carve (16B-aligned chunks)
  const size_t nd = (size_t)N * DH;
  float* dinv   = (float*)d_ws;
  float* H      = dinv + al4(N);
  float* AGG    = H + nd;
  float* X0     = AGG + nd;
  float* stats  = X0 + nd;                 // 256 floats
  int*   cnt    = (int*)(stats + 256);
  int*   offs   = cnt + al4(N);            // N+1 entries
  int*   cursor = offs + al4((size_t)N + 1);
  int*   srcIdx = cursor + al4(N);         // E entries

  // ---- CSR-by-destination + dinv (once per call) ----
  zero_kernel<<<cdiv_i(N, 256 * 8), 256, 0, stream>>>((float*)cnt, (long long)N);
  count_kernel<<<cdiv_i(E, 256), 256, 0, stream>>>(colI, cnt, E);
  scan_kernel<<<1, 1024, 0, stream>>>(cnt, offs, cursor, N);
  dinv_kernel<<<cdiv_i(N, 256), 256, 0, stream>>>(cnt, dinv, N);
  fill_kernel<<<cdiv_i(E, 256), 256, 0, stream>>>(rowI, colI, cursor, srcIdx, E);

  const int strips  = cdiv_i(N, 16);
  const int gblocks = cdiv_i(strips, 8);
  const int bnblk   = (int)((nd + 255) / 256);

  // ---- fc + BN0 + relu ; x0 = h ----
  gemm_kernel<128, 128><<<gblocks, 256, 0, stream>>>(x, fc_w, fc_b, H, N);
  zero_kernel<<<1, 256, 0, stream>>>(stats, 256);
  stats_kernel<<<1024, 256, 0, stream>>>(H, stats, N);
  bn_kernel<<<bnblk, 256, 0, stream>>>(H, stats, bn_gamma, bn_beta, nullptr, X0, N);

  // ---- 3x [gather-SPMM -> conv GEMM -> BN+relu+residual] ----
  for (int l = 0; l < 3; ++l) {
    spmm_gather_kernel<<<cdiv_i(N, 8), 256, 0, stream>>>(offs, srcIdx, dinv, H, AGG, N);
    gemm_kernel<128, 128><<<gblocks, 256, 0, stream>>>(
        AGG, conv_w + (size_t)l * DH * DH, conv_b + (size_t)l * DH, H, N);
    zero_kernel<<<1, 256, 0, stream>>>(stats, 256);
    stats_kernel<<<1024, 256, 0, stream>>>(H, stats, N);
    bn_kernel<<<bnblk, 256, 0, stream>>>(
        H, stats, bn_gamma + (size_t)(l + 1) * DH, bn_beta + (size_t)(l + 1) * DH,
        X0, nullptr, N);
  }

  // ---- classifier: 40 cols padded to 48 (3 WMMA column tiles) ----
  gemm_kernel<48, 40><<<gblocks, 256, 0, stream>>>(H, cls_w, cls_b, out, N);
}
